// Up1_13365938225164
// MI455X (gfx1250) — matrix operand order repaired and checked by reading, not compiled, not run on hardware
//
#include <hip/hip_runtime.h>

typedef __attribute__((ext_vector_type(16))) _Float16 v16h;
typedef __attribute__((ext_vector_type(8)))  float    v8f;

union H16 { v16h v; uint4 q[2]; };

#define NB_    8
#define HW_    4096
#define NPOS   32768
#define HP_    66
#define PLANE  4356      // 66*66
#define KDCN   1152      // 128*9
#define CHUNK  8192
#define EPS_   1e-5f

// ---------------- stage 1: upsample(x1) + concat(x2) + depthwise3x3 + bias -> h1 (pos,256) f16 ----
__device__ __forceinline__ float up2x_val(const float* __restrict__ x1, int b, int c, int y, int x) {
  const float s = 31.0f / 63.0f;
  float cy = (float)y * s, cx = (float)x * s;
  int iy = (int)cy; if (iy > 30) iy = 30;
  int ix = (int)cx; if (ix > 30) ix = 30;
  float ty = cy - (float)iy, tx = cx - (float)ix;
  const float* p = x1 + (((size_t)b * 128 + c) * 32 + iy) * 32 + ix;
  float r0 = p[0]  * (1.f - ty) + p[32] * ty;
  float r1 = p[1]  * (1.f - ty) + p[33] * ty;
  return r0 * (1.f - tx) + r1 * tx;
}

__device__ __forceinline__ float cat_val(const float* __restrict__ x1, const float* __restrict__ x2,
                                         int b, int c, int y, int x) {
  if ((unsigned)y >= 64u || (unsigned)x >= 64u) return 0.f;
  if (c < 128) return x2[(((size_t)b * 128 + c) * 64 + y) * 64 + x];
  return up2x_val(x1, b, c - 128, y, x);
}

__global__ void stage1_kernel(const float* __restrict__ x1, const float* __restrict__ x2,
                              const float* __restrict__ dww, const float* __restrict__ dwb,
                              _Float16* __restrict__ h1) {
  int t = blockIdx.x * blockDim.x + threadIdx.x;     // 8*256*64*64
  int w = t & 63, h = (t >> 6) & 63, c = (t >> 12) & 255, b = t >> 20;
  float acc = dwb[c];
  const float* wk = dww + c * 9;
  #pragma unroll
  for (int dy = -1; dy <= 1; ++dy)
    #pragma unroll
    for (int dx = -1; dx <= 1; ++dx)
      acc += wk[(dy + 1) * 3 + (dx + 1)] * cat_val(x1, x2, b, c, h + dy, w + dx);
  h1[((size_t)(b << 12) + (h << 6) + w) * 256 + c] = (_Float16)acc;
}

// ---------------- f32 -> f16 weight pack (layout already Bt[n][k]) ----------------
__global__ void pack_half_kernel(const float* __restrict__ src, _Float16* __restrict__ dst, int n) {
  int t = blockIdx.x * blockDim.x + threadIdx.x;
  if (t < n) dst[t] = (_Float16)src[t];
}

// ---------------- WMMA GEMM: out[pos,128] = A[M,K](f16,row-major) x Bt[128,K](f16) (+bias) ----
// Block = 128 threads = 4 waves; block M-tile = 64; each wave owns a 16-row strip and the
// full N=128 (8 accumulator tiles) so A fragments are loaded exactly once. B tiles are loaded
// per-WMMA (B is <=288 KB, L0-resident); each WMMA retires in the matrix pipe while the next
// tile's loads issue, so only the first tile's latency is exposed. Epilogue is pos-major:
// 8 stores per tile share one base address + immediate offsets (no address-math pressure).
template <int K, bool BIAS>
__global__ void gemm_f16_kernel(const _Float16* __restrict__ A, const _Float16* __restrict__ Bt,
                                const float* __restrict__ bias, float* __restrict__ out,
                                int m_base) {
  const int lane = threadIdx.x & 31;
  const int wave = threadIdx.x >> 5;          // 0..3
  const int hi   = (lane >> 4) & 1;
  const int m0   = blockIdx.x * 64 + wave * 16;   // local M strip for this wave
  const int arow = m0 + (lane & 15);
  const int akoff = hi * 8;
  const int bkoff = hi * 8;
  const int bcol  = lane & 15;

  v8f acc[8];
  #pragma unroll
  for (int nt = 0; nt < 8; ++nt) acc[nt] = (v8f){0.f,0.f,0.f,0.f,0.f,0.f,0.f,0.f};

  for (int k0 = 0; k0 < K; k0 += 32) {
    H16 a;
    const _Float16* ap = A + (size_t)arow * K + (k0 + akoff);
    a.q[0] = *(const uint4*)ap;
    a.q[1] = *(const uint4*)(ap + 16);
    #pragma unroll
    for (int nt = 0; nt < 8; ++nt) {
      H16 b;
      const _Float16* bp = Bt + (size_t)(nt * 16 + bcol) * K + (k0 + bkoff);
      b.q[0] = *(const uint4*)bp;
      b.q[1] = *(const uint4*)(bp + 16);
      acc[nt] = __builtin_amdgcn_wmma_f32_16x16x32_f16(false, a.v, false, b.v, (short)0,
                                                       acc[nt], false, false);
    }
  }

  // pos-major epilogue: row r of this lane-half maps to gpos = m_base + m0 + rowadd + r,
  // i.e. a single base pointer with r*128-float immediate offsets.
  const int rowadd = hi * 8;
  float* obase = out + (size_t)(m_base + m0 + rowadd) * 128 + bcol;
  #pragma unroll
  for (int nt = 0; nt < 8; ++nt) {
    float bv = BIAS ? bias[nt * 16 + bcol] : 0.f;
    float* op = obase + nt * 16;
    #pragma unroll
    for (int r = 0; r < 8; ++r) {
      op[(size_t)r * 128] = acc[nt][r] + bv;
    }
  }
}

// ---------------- scatter pos-major (pos,128) -> padded NCHW (B,128,66,66) interior ----------------
__global__ void scatter_pad_kernel(const float* __restrict__ src, float* __restrict__ dst) {
  int t = blockIdx.x * blockDim.x + threadIdx.x;      // 8*128*4096, NCHW order
  int hw = t & 4095, c = (t >> 12) & 127, b = t >> 19;
  float v = src[((size_t)(b << 12) + hw) * 128 + c];
  int h = hw >> 6, w = hw & 63;
  dst[(((size_t)b * 128 + c) * HP_ + h + 1) * HP_ + (w + 1)] = v;
}

// ---------------- offset/mask 3x3 conv (128 -> 27), direct VALU ----------------
__global__ void offmask_kernel(const float* __restrict__ h2p,
                               const float* __restrict__ p_w, const float* __restrict__ p_b,
                               const float* __restrict__ m_w, const float* __restrict__ m_b,
                               float* __restrict__ offb, float* __restrict__ mskb) {
  int t = blockIdx.x * blockDim.x + threadIdx.x;      // 8*27*4096
  int hw = t & 4095;
  int oc = (t >> 12) % 27;
  int b  = t / (27 * 4096);
  int h = hw >> 6, w = hw & 63;
  const float* wgt = (oc < 18) ? (p_w + oc * KDCN) : (m_w + (oc - 18) * KDCN);
  float acc = (oc < 18) ? p_b[oc] : m_b[oc - 18];
  for (int ic = 0; ic < 128; ++ic) {
    const float* base = h2p + (((size_t)b * 128 + ic) * HP_ + h) * HP_ + w;
    const float* wk = wgt + ic * 9;
    acc += wk[0]*base[0]        + wk[1]*base[1]        + wk[2]*base[2]
         + wk[3]*base[HP_]      + wk[4]*base[HP_+1]    + wk[5]*base[HP_+2]
         + wk[6]*base[2*HP_]    + wk[7]*base[2*HP_+1]  + wk[8]*base[2*HP_+2];
  }
  if (oc < 18) offb[((size_t)b * 18 + oc) * HW_ + hw] = acc;
  else         mskb[((size_t)b * 9 + (oc - 18)) * HW_ + hw] = 1.f / (1.f + expf(-acc));
}

// ---------------- per (pixel, tap): 4 bilinear weights*mod + 4 plane indices ----------------
__global__ void coords_kernel(const float* __restrict__ offb, const float* __restrict__ mskb,
                              float4* __restrict__ tapw, int4* __restrict__ tapi) {
  int t = blockIdx.x * blockDim.x + threadIdx.x;      // 32768*9
  int n = t % 9, pos = t / 9;
  int b = pos >> 12, hw = pos & 4095, h = hw >> 6, w = hw & 63;
  float ox = offb[((size_t)b * 18 + n) * HW_ + hw];
  float oy = offb[((size_t)b * 18 + 9 + n) * HW_ + hw];
  float mv = mskb[((size_t)b * 9 + n) * HW_ + hw];
  float px = (float)(h + 1) + (float)(n / 3 - 1) + ox;
  float py = (float)(w + 1) + (float)(n % 3 - 1) + oy;
  float fx = floorf(px), fy = floorf(py);
  float qx0 = fminf(fmaxf(fx,       0.f), 65.f);
  float qx1 = fminf(fmaxf(fx + 1.f, 0.f), 65.f);
  float qy0 = fminf(fmaxf(fy,       0.f), 65.f);
  float qy1 = fminf(fmaxf(fy + 1.f, 0.f), 65.f);
  float pxc = fminf(fmaxf(px, 0.f), 65.f);
  float pyc = fminf(fmaxf(py, 0.f), 65.f);
  float glt = (1.f + (qx0 - pxc)) * (1.f + (qy0 - pyc));
  float grb = (1.f - (qx1 - pxc)) * (1.f - (qy1 - pyc));
  float glb = (1.f + (qx0 - pxc)) * (1.f - (qy1 - pyc));
  float grt = (1.f - (qx1 - pxc)) * (1.f + (qy0 - pyc));
  int ix0 = (int)qx0, ix1 = (int)qx1, iy0 = (int)qy0, iy1 = (int)qy1;
  tapw[t] = make_float4(glt * mv, grb * mv, glb * mv, grt * mv);
  tapi[t] = make_int4(ix0 * HP_ + iy0, ix1 * HP_ + iy1, ix0 * HP_ + iy1, ix1 * HP_ + iy0);
}

// ---------------- gather im2col A3 chunk: A3[lpos][c*9+n] f16 ----------------
__global__ void gather_kernel(const float* __restrict__ h2p, const float4* __restrict__ tapw,
                              const int4* __restrict__ tapi, _Float16* __restrict__ A3, int pos_base) {
  int t = blockIdx.x * blockDim.x + threadIdx.x;      // 128*CHUNK
  int lpos = t & (CHUNK - 1);
  int c = t >> 13;
  int pos = pos_base + lpos;
  int b = pos >> 12;
  const float* base = h2p + ((size_t)b * 128 + c) * PLANE;
  _Float16* outp = A3 + (size_t)lpos * KDCN + c * 9;
  const float4* wp = tapw + (size_t)pos * 9;
  const int4*  ip = tapi + (size_t)pos * 9;
  #pragma unroll
  for (int n = 0; n < 9; ++n) {
    float4 wv = wp[n]; int4 iv = ip[n];
    float s = wv.x * base[iv.x] + wv.y * base[iv.y] + wv.z * base[iv.z] + wv.w * base[iv.w];
    outp[n] = (_Float16)s;
  }
}

// ---------------- BN stats: pos-major (pos,128) ----------------
__global__ void stats_posmajor_kernel(const float* __restrict__ x, float* __restrict__ mean,
                                      float* __restrict__ var) {
  int c = blockIdx.x;
  float s = 0.f, ss = 0.f;
  for (int i = threadIdx.x; i < NPOS; i += blockDim.x) {
    float v = x[(size_t)i * 128 + c]; s += v; ss += v * v;
  }
  __shared__ float sh[256], sh2[256];
  sh[threadIdx.x] = s; sh2[threadIdx.x] = ss; __syncthreads();
  for (int st = 128; st > 0; st >>= 1) {
    if ((int)threadIdx.x < st) { sh[threadIdx.x] += sh[threadIdx.x + st]; sh2[threadIdx.x] += sh2[threadIdx.x + st]; }
    __syncthreads();
  }
  if (threadIdx.x == 0) {
    float mu = sh[0] / (float)NPOS;
    mean[c] = mu; var[c] = sh2[0] / (float)NPOS - mu * mu;
  }
}

// ---------------- BN stats: NCHW (8,128,64,64) ----------------
__global__ void stats_nchw_kernel(const float* __restrict__ x, float* __restrict__ mean,
                                  float* __restrict__ var) {
  int c = blockIdx.x;
  float s = 0.f, ss = 0.f;
  for (int i = threadIdx.x; i < NB_ * HW_; i += blockDim.x) {
    int b = i >> 12, hw = i & 4095;
    float v = x[(((size_t)b * 128 + c) << 12) + hw]; s += v; ss += v * v;
  }
  __shared__ float sh[256], sh2[256];
  sh[threadIdx.x] = s; sh2[threadIdx.x] = ss; __syncthreads();
  for (int st = 128; st > 0; st >>= 1) {
    if ((int)threadIdx.x < st) { sh[threadIdx.x] += sh[threadIdx.x + st]; sh2[threadIdx.x] += sh2[threadIdx.x + st]; }
    __syncthreads();
  }
  if (threadIdx.x == 0) {
    float mu = sh[0] / (float)NPOS;
    mean[c] = mu; var[c] = sh2[0] / (float)NPOS - mu * mu;
  }
}

// ---------------- BN + exact GELU -> padded NCHW buffer ----------------
__global__ void bn_gelu_kernel(const float* __restrict__ dcn, const float* __restrict__ mean,
                               const float* __restrict__ var, const float* __restrict__ g,
                               const float* __restrict__ bb, float* __restrict__ g1p) {
  int t = blockIdx.x * blockDim.x + threadIdx.x;      // 8*128*4096
  int hw = t & 4095, c = (t >> 12) & 127, b = t >> 19;
  float v = dcn[((size_t)(b << 12) + hw) * 128 + c];
  float xn = (v - mean[c]) * rsqrtf(var[c] + EPS_) * g[c] + bb[c];
  float gl = 0.5f * xn * (1.f + erff(xn * 0.70710678118654752f));
  int h = hw >> 6, w = hw & 63;
  g1p[(((size_t)b * 128 + c) * HP_ + h + 1) * HP_ + (w + 1)] = gl;
}

// ---------------- depthwise 3x3 #2 ----------------
__global__ void dw2_kernel(const float* __restrict__ g1p, const float* __restrict__ wgt,
                           const float* __restrict__ bias, float* __restrict__ h3) {
  int t = blockIdx.x * blockDim.x + threadIdx.x;      // NCHW (8,128,64,64)
  int hw = t & 4095, c = (t >> 12) & 127, b = t >> 19;
  int h = hw >> 6, w = hw & 63;
  const float* base = g1p + (((size_t)b * 128 + c) * HP_ + h) * HP_ + w;
  const float* wk = wgt + c * 9;
  float acc = bias[c]
      + wk[0]*base[0]       + wk[1]*base[1]       + wk[2]*base[2]
      + wk[3]*base[HP_]     + wk[4]*base[HP_+1]   + wk[5]*base[HP_+2]
      + wk[6]*base[2*HP_]   + wk[7]*base[2*HP_+1] + wk[8]*base[2*HP_+2];
  h3[t] = acc;
}

// ---------------- BN + ReLU -> d_out ----------------
__global__ void bn_relu_kernel(const float* __restrict__ h3, const float* __restrict__ mean,
                               const float* __restrict__ var, const float* __restrict__ g,
                               const float* __restrict__ bb, float* __restrict__ out) {
  int t = blockIdx.x * blockDim.x + threadIdx.x;      // NCHW
  int c = (t >> 12) & 127;
  float xn = (h3[t] - mean[c]) * rsqrtf(var[c] + EPS_) * g[c] + bb[c];
  out[t] = fmaxf(xn, 0.f);
}

extern "C" void kernel_launch(void* const* d_in, const int* in_sizes, int n_in,
                              void* d_out, int out_size, void* d_ws, size_t ws_size,
                              hipStream_t stream) {
  (void)in_sizes; (void)n_in; (void)out_size; (void)ws_size;
  const float* x1     = (const float*)d_in[0];
  const float* x2     = (const float*)d_in[1];
  const float* dw1_w  = (const float*)d_in[2];
  const float* dw1_b  = (const float*)d_in[3];
  const float* pw_w   = (const float*)d_in[4];
  const float* pw_b   = (const float*)d_in[5];
  const float* p_w    = (const float*)d_in[6];
  const float* p_b    = (const float*)d_in[7];
  const float* m_w    = (const float*)d_in[8];
  const float* m_b    = (const float*)d_in[9];
  const float* dcn_w  = (const float*)d_in[10];
  const float* bn1_g  = (const float*)d_in[11];
  const float* bn1_b  = (const float*)d_in[12];
  const float* dw2_w  = (const float*)d_in[13];
  const float* dw2_b  = (const float*)d_in[14];
  const float* bn2_g  = (const float*)d_in[15];
  const float* bn2_b  = (const float*)d_in[16];

  char* ws = (char*)d_ws;
  size_t off = 0;
  auto take = [&](size_t bytes) -> char* {
    char* p = ws + off;
    off += (bytes + 255) & ~(size_t)255;
    return p;
  };

  _Float16* h1   = (_Float16*)take((size_t)NPOS * 256 * 2);             // 16.8 MB (aliased by dcn)
  float*    dcn  = (float*)h1;                                          // alias: h1 dead after pointwise GEMM
  float*    h2p  = (float*)take((size_t)NB_ * 128 * PLANE * 4);         // 17.8 MB (aliased by g1p)
  float*    g1p  = h2p;                                                 // alias: h2p dead after gathers
  _Float16* w1t  = (_Float16*)take((size_t)128 * 256 * 2);
  _Float16* bt3  = (_Float16*)take((size_t)128 * KDCN * 2);
  float*    offb = (float*)take((size_t)NB_ * 18 * HW_ * 4);
  float*    mskb = (float*)take((size_t)NB_ * 9 * HW_ * 4);
  float4*   tapw = (float4*)take((size_t)NPOS * 9 * 16);
  int4*     tapi = (int4*)take((size_t)NPOS * 9 * 16);
  _Float16* a3   = (_Float16*)take((size_t)CHUNK * KDCN * 2);           // 18.9 MB chunk
  float*    h2pm = (float*)a3;   // alias: pos-major pointwise result, dead before a3 first written
  float*    h3   = (float*)take((size_t)NB_ * 128 * HW_ * 4);
  float*    mean1 = (float*)take(512);
  float*    var1  = (float*)take(512);
  float*    mean2 = (float*)take(512);
  float*    var2  = (float*)take(512);

  // zero padded buffer once: interior is overwritten, border zeros serve h2p and (via alias) g1p
  hipMemsetAsync(h2p, 0, (size_t)NB_ * 128 * PLANE * 4, stream);

  stage1_kernel<<<(NB_ * 256 * HW_) / 256, 256, 0, stream>>>(x1, x2, dw1_w, dw1_b, h1);
  pack_half_kernel<<<(128 * 256 + 255) / 256, 256, 0, stream>>>(pw_w, w1t, 128 * 256);
  pack_half_kernel<<<(128 * KDCN + 255) / 256, 256, 0, stream>>>(dcn_w, bt3, 128 * KDCN);

  // pointwise 256->128 GEMM -> pos-major temp, then scatter into padded NCHW
  gemm_f16_kernel<256, true><<<NPOS / 64, 128, 0, stream>>>(h1, w1t, pw_b, h2pm, 0);
  scatter_pad_kernel<<<(NB_ * 128 * HW_) / 256, 256, 0, stream>>>(h2pm, h2p);

  offmask_kernel<<<(NB_ * 27 * HW_) / 256, 256, 0, stream>>>(h2p, p_w, p_b, m_w, m_b, offb, mskb);
  coords_kernel<<<(NPOS * 9) / 256, 256, 0, stream>>>(offb, mskb, tapw, tapi);

  // deformable einsum: chunked im2col gather + WMMA GEMM (M=8192, K=1152, N=128)
  for (int ch = 0; ch < NPOS / CHUNK; ++ch) {
    gather_kernel<<<(128 * CHUNK) / 256, 256, 0, stream>>>(h2p, tapw, tapi, a3, ch * CHUNK);
    gemm_f16_kernel<KDCN, false><<<CHUNK / 64, 128, 0, stream>>>(a3, bt3, nullptr, dcn, ch * CHUNK);
  }

  stats_posmajor_kernel<<<128, 256, 0, stream>>>(dcn, mean1, var1);
  bn_gelu_kernel<<<(NB_ * 128 * HW_) / 256, 256, 0, stream>>>(dcn, mean1, var1, bn1_g, bn1_b, g1p);
  dw2_kernel<<<(NB_ * 128 * HW_) / 256, 256, 0, stream>>>(g1p, dw2_w, dw2_b, h3);
  stats_nchw_kernel<<<128, 256, 0, stream>>>(h3, mean2, var2);
  bn_relu_kernel<<<(NB_ * 128 * HW_) / 256, 256, 0, stream>>>(h3, mean2, var2, bn2_g, bn2_b, (float*)d_out);
}
